// SpatialEdgesEmbedding_60644938219640
// MI455X (gfx1250) — matrix operand order repaired and checked by reading, not compiled, not run on hardware
//
#include <hip/hip_runtime.h>

// ---------------------------------------------------------------------------
// relu(flat[65536,72] @ W[1800,72]^T + b[1800])  -> out[65536,1800] f32
// M=65536, K=72, N=1800.  Output write (472 MB) dominates -> memory bound
// (~21 us floor at 23.3 TB/s).  f16 WMMA (v_wmma_f32_16x16x32_f16) with f32
// accumulate keeps compute (~17 GFLOP) well under that floor; f32 WMMA
// (16x16x4) would be compute bound.  K padded 72->96 (3 k-steps), N padded
// 1800->1808 (113 tiles).  W pre-packed into per-lane B-fragment layout in
// d_ws (f16, 347 KB, L2-resident).  Each block: 2 M-tiles x all N-tiles
// (8 waves stride the 113 N-tiles), amortizing each B-fragment load over two
// accumulators.
// ---------------------------------------------------------------------------

typedef __attribute__((ext_vector_type(16))) _Float16 v16h;
typedef __attribute__((ext_vector_type(8)))  float    v8f;

#define IN_F    72
#define OUT_F   1800
#define NTILES  113      // ceil(1800/16)
#define KSTEPS  3        // ceil(72/32)
#define WAVES   8        // waves per block
#define MT      2        // M-tiles per block

// ---------------------------------------------------------------------------
// Pack W[1800,72] f32 into f16 B-fragments, fragment-major:
//   Wp[((t*3+s)*32 + lane)*16 + i] = W[o=t*16+(lane&15)][k=s*32+(lane<16?0:8)+i+(i<8?0:8)]
// (B 32x16 16-bit layout: lane = column N, lanes 0-15 hold K 0-7,16-23 of the step,
//  lanes 16-31 hold K 8-15,24-31, matching the A fragments.)  Zero padding for
//  o >= 1800 and k >= 72.
// ---------------------------------------------------------------------------
__global__ __launch_bounds__(256)
void wpack_kernel(const float* __restrict__ W, _Float16* __restrict__ Wp, int total) {
    int idx = blockIdx.x * 256 + threadIdx.x;
    if (idx >= total) return;
    int i    = idx & 15;
    int l    = (idx >> 4) & 31;
    int frag = idx >> 9;
    int s    = frag % KSTEPS;
    int t    = frag / KSTEPS;
    int o    = t * 16 + (l & 15);
    int k    = s * 32 + ((l < 16) ? 0 : 8) + i + ((i < 8) ? 0 : 8);
    _Float16 v = (_Float16)0.0f;
    if (o < OUT_F && k < IN_F) v = (_Float16)W[o * IN_F + k];
    Wp[idx] = v;
}

__device__ __forceinline__ void pack4(v16h& a, int base, float4 f) {
    a[base + 0] = (_Float16)f.x;
    a[base + 1] = (_Float16)f.y;
    a[base + 2] = (_Float16)f.z;
    a[base + 3] = (_Float16)f.w;
}

__global__ __launch_bounds__(256)
void SpatialEdgesEmbedding_gemm_kernel(const float* __restrict__ x,
                                       const _Float16* __restrict__ Wp,
                                       const float* __restrict__ bias,
                                       float* __restrict__ out) {
    const int lane   = threadIdx.x & 31;
    // Force the wave id scalar: makes the N-tile loop counter, trip test and
    // prefetch guard SALU/s_cbranch instead of v_cmpx + EXEC juggling, and
    // lets B-fragment addresses use the SGPR-base form.
    const int wave   = __builtin_amdgcn_readfirstlane(threadIdx.x >> 5);
    const int mtile0 = blockIdx.x * MT;
    const int m      = lane & 15;
    const bool hi    = lane >= 16;
    const int kb4    = hi ? 2 : 0;   // per-lane K base (0 or 8 floats) in float4 units

    // ---- Build A fragments for MT m-tiles x 3 k-steps (16-bit A 16x32 layout:
    // lane L holds row M=L&15; lanes<16 K{b..b+7, b+16..b+23} with b=0,
    // lanes>=16 the same with b=8).  x rows are 72 contiguous f32 per chunk.
    v16h a[MT][KSTEPS];
    #pragma unroll
    for (int mt = 0; mt < MT; ++mt) {
        const float4* xv =
            (const float4*)(x + ((size_t)(mtile0 + mt) * 16 + m) * IN_F);
        v16h t0 = {};
        pack4(t0, 0,  xv[kb4 + 0]);  pack4(t0, 4,  xv[kb4 + 1]);
        pack4(t0, 8,  xv[kb4 + 4]);  pack4(t0, 12, xv[kb4 + 5]);
        a[mt][0] = t0;
        v16h t1 = {};
        pack4(t1, 0,  xv[kb4 + 8]);  pack4(t1, 4,  xv[kb4 + 9]);
        pack4(t1, 8,  xv[kb4 + 12]); pack4(t1, 12, xv[kb4 + 13]);
        a[mt][1] = t1;
        v16h t2 = {};                        // k-step 2 covers K 64..95: only 64..71 valid
        if (!hi) { pack4(t2, 0, xv[16]); pack4(t2, 4, xv[17]); }
        a[mt][2] = t2;
    }

    const v16h* bfrag = (const v16h*)Wp;
    const int col = lane & 15;

    for (int t = wave; t < NTILES; t += WAVES) {
        const v16h* bf = bfrag + (size_t)t * (KSTEPS * 32) + lane;
        v16h b0 = bf[0];
        v16h b1 = bf[32];
        v16h b2 = bf[64];

        // Unconditional bias load with clamped index: clauses with the B
        // loads and hides under the WMMAs instead of stalling after them.
        const int o  = t * 16 + col;
        const int oc = o < OUT_F ? o : (OUT_F - 1);
        const float bv = bias[oc];

        if (t + WAVES < NTILES)   // scalar branch: warm path for next tile
            __builtin_prefetch(bfrag + (size_t)(t + WAVES) * (KSTEPS * 32) + lane, 0, 1);

        v8f acc0 = {};
        v8f acc1 = {};
        // Interleave the two accumulators so consecutive WMMAs are independent.
        acc0 = __builtin_amdgcn_wmma_f32_16x16x32_f16(false, a[0][0], false, b0,
                                                      (short)0, acc0, false, false);
        acc1 = __builtin_amdgcn_wmma_f32_16x16x32_f16(false, a[1][0], false, b0,
                                                      (short)0, acc1, false, false);
        acc0 = __builtin_amdgcn_wmma_f32_16x16x32_f16(false, a[0][1], false, b1,
                                                      (short)0, acc0, false, false);
        acc1 = __builtin_amdgcn_wmma_f32_16x16x32_f16(false, a[1][1], false, b1,
                                                      (short)0, acc1, false, false);
        acc0 = __builtin_amdgcn_wmma_f32_16x16x32_f16(false, a[0][2], false, b2,
                                                      (short)0, acc0, false, false);
        acc1 = __builtin_amdgcn_wmma_f32_16x16x32_f16(false, a[1][2], false, b2,
                                                      (short)0, acc1, false, false);

        if (o < OUT_F) {
            // C/D layout: VGPR j -> row M = j + (hi?8:0), column = lane&15.
            const size_t base0 =
                ((size_t)mtile0 * 16 + (hi ? 8 : 0)) * OUT_F + (size_t)o;
            #pragma unroll
            for (int j = 0; j < 8; ++j) {
                float v0 = acc0[j] + bv;
                v0 = v0 > 0.0f ? v0 : 0.0f;
                __builtin_nontemporal_store(v0, out + base0 + (size_t)j * OUT_F);
                float v1 = acc1[j] + bv;
                v1 = v1 > 0.0f ? v1 : 0.0f;
                __builtin_nontemporal_store(
                    v1, out + base0 + (size_t)(16 * OUT_F) + (size_t)j * OUT_F);
            }
        }
    }
}

extern "C" void kernel_launch(void* const* d_in, const int* in_sizes, int n_in,
                              void* d_out, int out_size, void* d_ws, size_t ws_size,
                              hipStream_t stream) {
    const float* x  = (const float*)d_in[0];   // [N,6,3] f32 == [chunks,72]
    const float* W  = (const float*)d_in[1];   // [1800,72] f32
    const float* b  = (const float*)d_in[2];   // [1800] f32
    float*       out = (float*)d_out;          // [chunks,1800] f32
    _Float16*    Wp  = (_Float16*)d_ws;        // packed B fragments (f16)

    const int chunks = in_sizes[0] / IN_F;     // 65536
    const int mtiles = chunks / 16;            // 4096
    const int blocks = mtiles / MT;            // 2048

    const int total  = NTILES * KSTEPS * 32 * 16;   // 173568 f16 fragment elems
    wpack_kernel<<<(total + 255) / 256, 256, 0, stream>>>(W, Wp, total);
    SpatialEdgesEmbedding_gemm_kernel<<<blocks, 256, 0, stream>>>(x, Wp, b, out);
}
